// LlamaAttentionWeightsOnly_11020886082212
// MI455X (gfx1250) — matrix operand order repaired and checked by reading, not compiled, not run on hardware
//
#include <hip/hip_runtime.h>

// ---------------------------------------------------------------------------
// LLaMA attention weights (QK^T + softmax), MI455X / gfx1250, wave32.
//   B=1, S=2048, D=2048, HQ=32, HKV=8, HD=64, EXPAND=4
//
// Pipeline:
//   K0: one-shot fp32->bf16 convert of H, Wq, Wk into workspace (each element
//       converted exactly once; native cvt instructions, 8 elems/thread).
//   K1: Q = H*Wq^T, K = H*Wk^T in bf16 WMMA (f32 accum), RoPE fused, bf16 out.
//       Inner loop is pure global_load_b128 + v_wmma_f32_16x16x32_bf16.
//   K2: per (head, 16-query tile): scores = Q*K^T via WMMA, staged in 128KB
//       LDS, row softmax (float4-vectorized passes), single coalesced b128
//       fp32 write of the 537MB output (HBM-bound: ~23us floor at 23.3 TB/s).
// All GEMM operands (8MB H + 8MB Wq + 2MB Wk + 8MB Qb + 2MB Kb) are
// L2-resident (192MB), so operand re-reads never touch HBM.
// ---------------------------------------------------------------------------

typedef __attribute__((ext_vector_type(16))) __bf16 v16bf;
typedef __attribute__((ext_vector_type(8)))  __bf16 v8bf;
typedef __attribute__((ext_vector_type(8)))  float  v8f;
typedef __attribute__((ext_vector_type(4)))  float  v4f;

#define S_  2048
#define D_  2048
#define HQ_ 32
#define HKV_ 8
#define HD_ 64
#define NH_ (HQ_ + HKV_)   // 40 projection "heads" (32 Q + 8 K)

// ---------------------------------------------------------------------------
// Kernel 0: fp32 -> bf16 bulk convert (native cvt path), 8 elements/thread.
// ---------------------------------------------------------------------------
__global__ __launch_bounds__(256) void cvt_bf16_kernel(
    const float* __restrict__ src, __bf16* __restrict__ dst, int n)
{
    int i = (blockIdx.x * 256 + threadIdx.x) * 8;
    if (i + 7 < n) {
        v4f a = *(const v4f*)(src + i);
        v4f b = *(const v4f*)(src + i + 4);
        v8bf r;
        #pragma unroll
        for (int j = 0; j < 4; ++j) {
            r[j]     = (__bf16)a[j];
            r[4 + j] = (__bf16)b[j];
        }
        *(v8bf*)(dst + i) = r;
    }
}

// ---------------------------------------------------------------------------
// Kernel 1: Q/K projection + fused RoPE, all-bf16 operands.
// One wave per (16-row s-tile, head). Wave computes a 16x64 slab via 4 WMMA
// accumulators so the full head dim lives in registers for rotate-half.
// ---------------------------------------------------------------------------
__global__ __launch_bounds__(128) void proj_rope_kernel(
    const __bf16* __restrict__ Hb,     // [S_][D_] bf16
    const float*  __restrict__ cosp,
    const float*  __restrict__ sinp,
    const __bf16* __restrict__ Wqb,    // [HQ_*HD_][D_] bf16
    const __bf16* __restrict__ Wkb,    // [HKV_*HD_][D_] bf16
    __bf16* __restrict__ Qb,           // [S_][HQ_*HD_] bf16
    __bf16* __restrict__ Kb)           // [S_][HKV_*HD_] bf16
{
    const int lane = threadIdx.x & 31;
    const int wave = threadIdx.x >> 5;
    const int gw   = blockIdx.x * 4 + wave;       // global wave id
    const int head  = gw % NH_;                   // 0..31 = Q head, 32..39 = K head
    const int mtile = gw / NH_;                   // 0..127 (s-tile)

    const bool isK = (head >= HQ_);
    const __bf16* W = isK ? Wkb : Wqb;
    const int hloc  = isK ? (head - HQ_) : head;
    const int obase = hloc * HD_;                 // row base into W, col base into dst
    const int s0    = mtile * 16;

    const int m  = lane & 15;    // A row within tile / B column within tile
    const int hh = lane >> 4;    // half-wave selector

    v8f acc[4] = {};             // 4 x (16x16 f32) = 16 rows x 64 head dims

    const __bf16* hrow = Hb + (size_t)(s0 + m) * D_;
    for (int kk = 0; kk < D_; kk += 32) {
        // A fragment: 16x32 bf16; lane half hh holds K = kk+8hh+{0..7} and
        // kk+16+8hh+{0..7}: two contiguous v8bf (16B) loads.
        v16bf a;
        {
            const __bf16* hp = hrow + kk + 8 * hh;
            v8bf lo = *(const v8bf*)(hp);
            v8bf hi = *(const v8bf*)(hp + 16);
            #pragma unroll
            for (int i = 0; i < 8; ++i) { a[i] = lo[i]; a[8 + i] = hi[i]; }
        }

        #pragma unroll
        for (int t = 0; t < 4; ++t) {
            // B fragment: B[k][n] = W[obase+16t+n][k]; lane holds K = kk+8hh+{0..7}
            // and kk+16+8hh+{0..7} of one W row -> two v8bf loads.
            const __bf16* wp = W + (size_t)(obase + t * 16 + m) * D_ + kk + 8 * hh;
            v16bf b;
            {
                v8bf lo = *(const v8bf*)(wp);
                v8bf hi = *(const v8bf*)(wp + 16);
                #pragma unroll
                for (int i = 0; i < 8; ++i) { b[i] = lo[i]; b[8 + i] = hi[i]; }
            }
            acc[t] = __builtin_amdgcn_wmma_f32_16x16x32_bf16(
                false, a, false, b, (short)0, acc[t], false, false);
        }
    }

    // Fused RoPE + bf16 store. C/D layout: lane m = col n, row = r + 8*hh.
    // Column j = 16t + n; rotate-half partner is tile t^2 at same lane/reg.
    const int ldDst = isK ? (HKV_ * HD_) : (HQ_ * HD_);
    __bf16* dst = isK ? Kb : Qb;
    #pragma unroll
    for (int t = 0; t < 4; ++t) {
        const int j = t * 16 + m;
        #pragma unroll
        for (int r = 0; r < 8; ++r) {
            const int s = s0 + r + 8 * hh;
            const float cv = cosp[s * HD_ + j];
            const float sv = sinp[s * HD_ + j];
            const float x   = acc[t][r];
            const float rot = (t < 2) ? -acc[t + 2][r] : acc[t - 2][r];
            dst[(size_t)s * ldDst + obase + j] = (__bf16)(x * cv + rot * sv);
        }
    }
}

// ---------------------------------------------------------------------------
// Kernel 2: per (head, 16-query tile): scores = Qh * Kh^T (bf16 WMMA, K-dim 64),
// staged in 16x2048 fp32 LDS (128KB dynamic, 2 blocks per 320KB WGP), then row
// softmax (float4 passes) + coalesced b128 output write. GQA: head h -> kv h/4.
// ---------------------------------------------------------------------------
__global__ __launch_bounds__(256) void scores_softmax_kernel(
    const __bf16* __restrict__ Qb,
    const __bf16* __restrict__ Kb,
    float* __restrict__ out)
{
    extern __shared__ float sS[];                 // [16][2048] scores

    const int lane = threadIdx.x & 31;
    const int wave = threadIdx.x >> 5;            // 0..7
    const int head  = blockIdx.x >> 7;            // /128
    const int qtile = blockIdx.x & 127;
    const int q0 = qtile * 16;
    const int kv = head >> 2;

    const int m  = lane & 15;
    const int hh = lane >> 4;

    // A fragments: Q rows q0..q0+15, head dims 0..63 as two 16x32 tiles.
    const __bf16* qp = Qb + (size_t)(q0 + m) * (HQ_ * HD_) + head * HD_;
    v16bf a0, a1;
    {
        v8bf lo = *(const v8bf*)(qp + 8 * hh);
        v8bf hi = *(const v8bf*)(qp + 16 + 8 * hh);
        #pragma unroll
        for (int i = 0; i < 8; ++i) { a0[i] = lo[i]; a0[8 + i] = hi[i]; }
        lo = *(const v8bf*)(qp + 32 + 8 * hh);
        hi = *(const v8bf*)(qp + 48 + 8 * hh);
        #pragma unroll
        for (int i = 0; i < 8; ++i) { a1[i] = lo[i]; a1[8 + i] = hi[i]; }
    }

    // Each wave computes 16 of the 128 16x16 score column-tiles.
    for (int it = 0; it < 16; ++it) {
        const int ct = wave + 8 * it;             // column tile 0..127
        const __bf16* kp = Kb + (size_t)(ct * 16 + m) * (HKV_ * HD_) + kv * HD_;
        if (it + 1 < 16) {
            // gfx1250 global_prefetch for the next K tile (L2-resident).
            __builtin_prefetch(Kb + (size_t)((ct + 8) * 16 + m) * (HKV_ * HD_)
                                  + kv * HD_, 0, 3);
        }
        // B[k][n] = K[ct*16+n][k]; lane holds head-dims 8hh+{0..7} and
        // 16+8hh+{0..7} of one key row -> two v8bf loads per fragment.
        v16bf b0, b1;
        {
            v8bf x0 = *(const v8bf*)(kp + 8 * hh);
            v8bf x1 = *(const v8bf*)(kp + 16 + 8 * hh);
            #pragma unroll
            for (int i = 0; i < 8; ++i) { b0[i] = x0[i]; b0[8 + i] = x1[i]; }
            v8bf x2 = *(const v8bf*)(kp + 32 + 8 * hh);
            v8bf x3 = *(const v8bf*)(kp + 48 + 8 * hh);
            #pragma unroll
            for (int i = 0; i < 8; ++i) { b1[i] = x2[i]; b1[8 + i] = x3[i]; }
        }
        v8f c = {};
        c = __builtin_amdgcn_wmma_f32_16x16x32_bf16(false, a0, false, b0,
                                                    (short)0, c, false, false);
        c = __builtin_amdgcn_wmma_f32_16x16x32_bf16(false, a1, false, b1,
                                                    (short)0, c, false, false);
        #pragma unroll
        for (int r = 0; r < 8; ++r)
            sS[(r + 8 * hh) * S_ + ct * 16 + m] = c[r];
    }

    __syncthreads();

    // Softmax: each wave owns rows 2*wave and 2*wave+1; all passes float4.
    const int c0 = lane * 4;                      // lane's base column
    #pragma unroll
    for (int rr = 0; rr < 2; ++rr) {
        const int row = wave * 2 + rr;
        float* rowp = sS + row * S_;

        // Pass 1: row max (v4 loads, 16 iters).
        float mx = -3.402823466e38f;
        for (int c = c0; c < S_; c += 128) {
            v4f v = *(const v4f*)(rowp + c);
            mx = fmaxf(mx, fmaxf(fmaxf(v[0], v[1]), fmaxf(v[2], v[3])));
        }
        #pragma unroll
        for (int off = 16; off > 0; off >>= 1)
            mx = fmaxf(mx, __shfl_xor(mx, off, 32));

        // Pass 2: exp in place + row sum.
        float sum = 0.0f;
        for (int c = c0; c < S_; c += 128) {
            v4f v = *(const v4f*)(rowp + c);
            v4f e;
            #pragma unroll
            for (int j = 0; j < 4; ++j) e[j] = __expf(v[j] - mx);
            *(v4f*)(rowp + c) = e;
            sum += (e[0] + e[1]) + (e[2] + e[3]);
        }
        #pragma unroll
        for (int off = 16; off > 0; off >>= 1)
            sum += __shfl_xor(sum, off, 32);
        const float inv = 1.0f / sum;

        // Pass 3: scale + coalesced b128 store (the HBM-bound write).
        float* op = out + ((size_t)head * S_ + (q0 + row)) * S_;
        for (int c = c0; c < S_; c += 128) {
            v4f v = *(const v4f*)(rowp + c);
            v4f o;
            #pragma unroll
            for (int j = 0; j < 4; ++j) o[j] = v[j] * inv;
            *(v4f*)(op + c) = o;
        }
    }
}

// ---------------------------------------------------------------------------
extern "C" void kernel_launch(void* const* d_in, const int* in_sizes, int n_in,
                              void* d_out, int out_size, void* d_ws, size_t ws_size,
                              hipStream_t stream) {
    const float* H    = (const float*)d_in[0];   // (1, 2048, 2048)
    const float* cosp = (const float*)d_in[1];   // (1, 2048, 64)
    const float* sinp = (const float*)d_in[2];   // (1, 2048, 64)
    // d_in[3] = attn_mask: all zeros and unused by the reference computation.
    const float* Wq   = (const float*)d_in[4];   // (2048, 2048)
    const float* Wk   = (const float*)d_in[5];   // (512, 2048)

    // Workspace layout (bf16): Hb 8MB | Wqb 8MB | Wkb 2MB | Qb 8MB | Kb 2MB
    __bf16* Hb  = (__bf16*)d_ws;
    __bf16* Wqb = Hb  + (size_t)S_ * D_;
    __bf16* Wkb = Wqb + (size_t)(HQ_ * HD_) * D_;
    __bf16* Qb  = Wkb + (size_t)(HKV_ * HD_) * D_;
    __bf16* Kb  = Qb  + (size_t)S_ * (HQ_ * HD_);

    // K0: bulk fp32->bf16 conversion (each element exactly once).
    const int nH  = S_ * D_;             // 4M
    const int nWq = HQ_ * HD_ * D_;      // 4M
    const int nWk = HKV_ * HD_ * D_;     // 1M
    cvt_bf16_kernel<<<nH  / (256 * 8), 256, 0, stream>>>(H,  Hb,  nH);
    cvt_bf16_kernel<<<nWq / (256 * 8), 256, 0, stream>>>(Wq, Wqb, nWq);
    cvt_bf16_kernel<<<nWk / (256 * 8), 256, 0, stream>>>(Wk, Wkb, nWk);

    // K1: 128 s-tiles * 40 heads = 5120 waves, 4 waves/block.
    const int blocksA = (128 * NH_) / 4;   // 1280
    proj_rope_kernel<<<blocksA, 128, 0, stream>>>(Hb, cosp, sinp, Wqb, Wkb, Qb, Kb);

    // K2: 32 heads * 128 query tiles, 128KB dynamic LDS per block.
    const int blocksB = HQ_ * 128;         // 4096
    const size_t ldsB = (size_t)16 * S_ * sizeof(float);   // 131072 B
    scores_softmax_kernel<<<blocksB, 256, ldsB, stream>>>(Qb, Kb, (float*)d_out);
}
